// RSTBWithInputConv_9964324127344
// MI455X (gfx1250) — matrix operand order repaired and checked by reading, not compiled, not run on hardware
//
#include <hip/hip_runtime.h>
#include <hip/hip_bf16.h>

typedef __bf16 bf16_t;
typedef __attribute__((ext_vector_type(16))) __bf16 v16bf;
typedef __attribute__((ext_vector_type(8)))  __bf16 v8bf;
typedef __attribute__((ext_vector_type(8)))  float  v8f;

#define B_SZ 2
#define D_SZ 6
#define CIN  3
#define H_SZ 96
#define W_SZ 96
#define CH   144
#define KP   160      // CH padded to multiple of 32 for WMMA K
#define HID  288      // already multiple of 32
#define HEADS 6
#define HD   24
#define HDP  32       // head dim padded for WMMA K
#define WN   128      // tokens per window (2*8*8)
#define NWIN 864      // total windows = B * 3 * 12 * 12
#define TOK  110592   // B*D*H*W
#define TBL  675

// ---------- helpers ----------
__device__ __forceinline__ bf16_t f2bf(float x) {
  union { float f; unsigned u; } a; a.f = x;
  unsigned r = a.u + 0x7fffu + ((a.u >> 16) & 1u);
  union { unsigned short s; bf16_t b; } o; o.s = (unsigned short)(r >> 16);
  return o.b;
}

__device__ __forceinline__ v16bf mkfrag(v8bf lo, v8bf hi) {
  v16bf r;
#pragma unroll
  for (int e = 0; e < 8; ++e) { r[e] = lo[e]; r[e + 8] = hi[e]; }
  return r;
}

// async global->LDS copy of one 16-byte chunk (CDNA5 ASYNCcnt path)
__device__ __forceinline__ void async_b128(const bf16_t* gsrc, bf16_t* ldst) {
  unsigned lds_off = (unsigned)(size_t)(void*)ldst;      // low 32 bits = LDS offset
  unsigned long long ga = (unsigned long long)(size_t)gsrc;
  asm volatile("global_load_async_to_lds_b128 %0, %1, off"
               :: "v"(lds_off), "v"(ga) : "memory");
}
__device__ __forceinline__ void async_wait() {
  asm volatile("s_wait_asynccnt 0" ::: "memory");
}

// ---------- conv(1x3x3) + bias + LN0 -> y0 and f ----------
__global__ __launch_bounds__(256) void conv_ln0_k(
    const float* __restrict__ x, const float* __restrict__ cw, const float* __restrict__ cb,
    const float* __restrict__ g, const float* __restrict__ bb,
    float* __restrict__ y0, float* __restrict__ f)
{
  int wave = threadIdx.x >> 5, lane = threadIdx.x & 31;
  int tok = blockIdx.x * 8 + wave;
  int b = tok / (D_SZ * H_SZ * W_SZ);
  int r = tok % (D_SZ * H_SZ * W_SZ);
  int d = r / (H_SZ * W_SZ);
  int h = (r / W_SZ) % H_SZ;
  int w = r % W_SZ;

  float xv[27];
  int j = 0;
#pragma unroll
  for (int ci = 0; ci < CIN; ++ci)
#pragma unroll
    for (int kh = 0; kh < 3; ++kh)
#pragma unroll
      for (int kw = 0; kw < 3; ++kw) {
        int hh = h + kh - 1, ww = w + kw - 1;
        float v = 0.f;
        if (hh >= 0 && hh < H_SZ && ww >= 0 && ww < W_SZ)
          v = x[(((size_t)(b * D_SZ + d) * CIN + ci) * H_SZ + hh) * W_SZ + ww];
        xv[j++] = v;
      }

  float vals[5]; float s = 0.f, ss = 0.f;
#pragma unroll
  for (int it = 0; it < 5; ++it) {
    int ch = lane + it * 32;
    float acc = 0.f;
    if (ch < CH) {
      acc = cb[ch];
      const float* wrow = cw + ch * 27;
#pragma unroll
      for (int k = 0; k < 27; ++k) acc += xv[k] * wrow[k];
      s += acc; ss += acc * acc;
    }
    vals[it] = acc;
  }
  for (int m = 16; m >= 1; m >>= 1) { s += __shfl_xor(s, m, 32); ss += __shfl_xor(ss, m, 32); }
  float mu = s / CH;
  float rs = rsqrtf(ss / CH - mu * mu + 1e-5f);
#pragma unroll
  for (int it = 0; it < 5; ++it) {
    int ch = lane + it * 32;
    if (ch < CH) {
      float o = (vals[it] - mu) * rs * g[ch] + bb[ch];
      y0[(size_t)tok * CH + ch] = o;
      f[(size_t)tok * CH + ch] = o;
    }
  }
}

// ---------- LayerNorm (+optional shift+window-partition) -> bf16 rows, K padded ----------
__global__ __launch_bounds__(256) void ln_part_k(
    const float* __restrict__ f, const float* __restrict__ g, const float* __restrict__ bb,
    bf16_t* __restrict__ out, int windowed, int shifted)
{
  int wave = threadIdx.x >> 5, lane = threadIdx.x & 31;
  int dst = blockIdx.x * 8 + wave;
  int src;
  if (windowed) {
    int win = dst >> 7, t = dst & 127;
    int b = win / 432, rr = win % 432;
    int wdb = rr / 144, r2 = rr % 144, whb = r2 / 12, wwb = r2 % 12;
    int zd = t >> 6, zy = (t >> 3) & 7, zx = t & 7;
    int sd = wdb * 2 + zd, sh = whb * 8 + zy, sw = wwb * 8 + zx;
    int od = sd, oh = sh, ow = sw;
    if (shifted) { od = (sd + 1) % D_SZ; oh = (sh + 4) % H_SZ; ow = (sw + 4) % W_SZ; }
    src = ((b * D_SZ + od) * H_SZ + oh) * W_SZ + ow;
  } else {
    src = dst;
  }
  const float* row = f + (size_t)src * CH;
  float vals[5]; float s = 0.f, ss = 0.f;
#pragma unroll
  for (int it = 0; it < 5; ++it) {
    int ch = lane + it * 32; float v = 0.f;
    if (ch < CH) { v = row[ch]; s += v; ss += v * v; }
    vals[it] = v;
  }
  for (int m = 16; m >= 1; m >>= 1) { s += __shfl_xor(s, m, 32); ss += __shfl_xor(ss, m, 32); }
  float mu = s / CH;
  float rs = rsqrtf(ss / CH - mu * mu + 1e-5f);
  bf16_t* orow = out + (size_t)dst * KP;
#pragma unroll
  for (int it = 0; it < 5; ++it) {
    int ch = lane + it * 32;
    if (ch < CH)      orow[ch] = f2bf((vals[it] - mu) * rs * g[ch] + bb[ch]);
    else if (ch < KP) orow[ch] = f2bf(0.f);
  }
}

// ---------- generic WMMA GEMM: O[M,N] = A[M,K] * W[N,K]^T, fused epilogues ----------
// Weight panel (48 x K) staged once per block into LDS via async-DMA (ASYNCcnt),
// removing the 8x per-wave redundant global fetch of B fragments.
// mode 0: fOut[row*ldOut+col] = val
// mode 1: qkv split -> q (scaled, +bias), k (+bias), vT (+bias, transposed), all bf16
// mode 2: bfOut = gelu(val + bias)
// mode 3: fOut[row*ldOut+col] += val + bias
__global__ __launch_bounds__(256) void gemm_wmma_k(
    const bf16_t* __restrict__ A, int lda,
    const bf16_t* __restrict__ Wt, int K, int N,
    const float* __restrict__ bias,
    int mode, float scale,
    float* __restrict__ fOut, int ldOut,
    bf16_t* __restrict__ bfOut, int ldBf,
    bf16_t* __restrict__ q, bf16_t* __restrict__ kk, bf16_t* __restrict__ vT)
{
  __shared__ __align__(32) bf16_t Bs[48 * HID];   // max 48*288*2 = 27.6 KB

  int wave = threadIdx.x >> 5, lane = threadIdx.x & 31;
  int g = lane >> 4, c = lane & 15;
  int m0 = blockIdx.x * 128 + wave * 16;
  int n0 = blockIdx.y * 48;

  // async-stage the 48-row weight panel for this block
  {
    const bf16_t* wpanel = Wt + (size_t)n0 * K;
    int chunks = 48 * K / 8;               // 16-byte chunks
    for (int idx = threadIdx.x; idx < chunks; idx += 256)
      async_b128(wpanel + (size_t)idx * 8, &Bs[(size_t)idx * 8]);
    async_wait();
    __syncthreads();
  }

  v8f acc0 = {}, acc1 = {}, acc2 = {};
  const bf16_t* arow = A + (size_t)(m0 + c) * lda;
  for (int k0 = 0; k0 < K; k0 += 32) {
    v8bf lo = *(const v8bf*)(arow + k0 + 16 * g);
    v8bf hi = *(const v8bf*)(arow + k0 + 16 * g + 8);
    v16bf af = mkfrag(lo, hi);
    const bf16_t* wbase = Bs + (size_t)c * K + k0 + 16 * g;
    v16bf b0 = *(const v16bf*)(wbase);
    v16bf b1 = *(const v16bf*)(wbase + (size_t)16 * K);
    v16bf b2 = *(const v16bf*)(wbase + (size_t)32 * K);
    acc0 = __builtin_amdgcn_wmma_f32_16x16x32_bf16(false, af, false, b0, (short)0, acc0, false, false);
    acc1 = __builtin_amdgcn_wmma_f32_16x16x32_bf16(false, af, false, b1, (short)0, acc1, false, false);
    acc2 = __builtin_amdgcn_wmma_f32_16x16x32_bf16(false, af, false, b2, (short)0, acc2, false, false);
  }

  v8f accs[3] = {acc0, acc1, acc2};
#pragma unroll
  for (int t = 0; t < 3; ++t) {
#pragma unroll
    for (int r = 0; r < 8; ++r) {
      int row = m0 + g * 8 + r;
      int col = n0 + t * 16 + c;
      float val = accs[t][r];
      if (mode == 0) {
        fOut[(size_t)row * ldOut + col] = val;
      } else if (mode == 1) {
        val += bias[col];
        int part = col / CH, rem = col % CH;
        int head = rem / HD, dd = rem % HD;
        int win = row >> 7, tt = row & 127;
        size_t bidx = ((size_t)(win * HEADS + head) * WN + tt) * HDP + dd;
        if (part == 0)      q[bidx] = f2bf(val * scale);
        else if (part == 1) kk[bidx] = f2bf(val);
        else                vT[((size_t)(win * HEADS + head) * HDP + dd) * WN + tt] = f2bf(val);
      } else if (mode == 2) {
        val += bias[col];
        float ge = 0.5f * val * (1.f + erff(val * 0.70710678118f));
        bfOut[(size_t)row * ldBf + col] = f2bf(ge);
      } else {
        fOut[(size_t)row * ldOut + col] += val + bias[col];
      }
    }
  }
}

// ---------- windowed attention: one block per (window, head) ----------
__global__ __launch_bounds__(256) void attn_k(
    const bf16_t* __restrict__ q, const bf16_t* __restrict__ kk, const bf16_t* __restrict__ vT,
    const float* __restrict__ rpb, bf16_t* __restrict__ ao, int shifted)
{
  __shared__ __align__(32) bf16_t P[WN * WN];     // 32 KB probabilities
  __shared__ __align__(32) bf16_t Ks[WN * HDP];   // 8 KB K panel
  __shared__ __align__(32) bf16_t Vs[HDP * WN];   // 8 KB V^T panel

  int wave = threadIdx.x >> 5, lane = threadIdx.x & 31;
  int g = lane >> 4, c = lane & 15;
  int win = blockIdx.x / HEADS, head = blockIdx.x % HEADS;
  const bf16_t* qh = q  + (size_t)(win * HEADS + head) * WN * HDP;
  const bf16_t* kh = kk + (size_t)(win * HEADS + head) * WN * HDP;
  const bf16_t* vh = vT + (size_t)(win * HEADS + head) * HDP * WN;
  int m0 = wave * 16;

  // async-stage K and V^T panels (shared by all 8 waves)
  {
    for (int idx = threadIdx.x; idx < WN * HDP / 8; idx += 256) {
      async_b128(kh + (size_t)idx * 8, &Ks[(size_t)idx * 8]);
      async_b128(vh + (size_t)idx * 8, &Vs[(size_t)idx * 8]);
    }
    async_wait();
    __syncthreads();
  }

  v8f zero = {};
  v8f s[8];
#pragma unroll
  for (int t = 0; t < 8; ++t) s[t] = zero;

  // S = Q * K^T   (K dim = 32 padded head dim)
  {
    const bf16_t* arow = qh + (size_t)(m0 + c) * HDP;
    v16bf af = mkfrag(*(const v8bf*)(arow + 16 * g), *(const v8bf*)(arow + 16 * g + 8));
#pragma unroll
    for (int t = 0; t < 8; ++t) {
      v16bf bfr = *(const v16bf*)(Ks + (size_t)(t * 16 + c) * HDP + 16 * g);
      s[t] = __builtin_amdgcn_wmma_f32_16x16x32_bf16(false, af, false, bfr, (short)0, s[t], false, false);
    }
  }

  // + relative position bias, + shift mask (computed on the fly)
  int wb = win % 432;
  int wdb = wb / 144, r2 = wb % 144, whb = r2 / 12, wwb = r2 % 12;
#pragma unroll
  for (int t = 0; t < 8; ++t) {
#pragma unroll
    for (int r = 0; r < 8; ++r) {
      int m = m0 + g * 8 + r;
      int n = t * 16 + c;
      int zm = m >> 6, ym = (m >> 3) & 7, xm = m & 7;
      int zn = n >> 6, yn = (n >> 3) & 7, xn = n & 7;
      int idx = ((zm - zn + 1) * 15 + (ym - yn + 7)) * 15 + (xm - xn + 7);
      float v = s[t][r] + rpb[idx * HEADS + head];
      if (shifted) {
        int gdm = wdb * 2 + zm, ghm = whb * 8 + ym, gwm = wwb * 8 + xm;
        int gdn = wdb * 2 + zn, ghn = whb * 8 + yn, gwn = wwb * 8 + xn;
        int lm  = (gdm < 4 ? 0 : (gdm < 5 ? 1 : 2)) * 9 +
                  (ghm < 88 ? 0 : (ghm < 92 ? 1 : 2)) * 3 +
                  (gwm < 88 ? 0 : (gwm < 92 ? 1 : 2));
        int ln_ = (gdn < 4 ? 0 : (gdn < 5 ? 1 : 2)) * 9 +
                  (ghn < 88 ? 0 : (ghn < 92 ? 1 : 2)) * 3 +
                  (gwn < 88 ? 0 : (gwn < 92 ? 1 : 2));
        if (lm != ln_) v -= 100.f;
      }
      s[t][r] = v;
    }
  }

  // softmax over the 128 columns of each row (8 tiles x 16 lanes of one half-wave)
#pragma unroll
  for (int r = 0; r < 8; ++r) {
    float mx = -1e30f;
#pragma unroll
    for (int t = 0; t < 8; ++t) mx = fmaxf(mx, s[t][r]);
    for (int msk = 8; msk >= 1; msk >>= 1) mx = fmaxf(mx, __shfl_xor(mx, msk, 32));
    float ex[8]; float sum = 0.f;
#pragma unroll
    for (int t = 0; t < 8; ++t) { ex[t] = __expf(s[t][r] - mx); sum += ex[t]; }
    for (int msk = 8; msk >= 1; msk >>= 1) sum += __shfl_xor(sum, msk, 32);
    float inv = 1.f / sum;
#pragma unroll
    for (int t = 0; t < 8; ++t) s[t][r] = ex[t] * inv;
  }

  // stage P as bf16 in LDS for the P@V WMMA
#pragma unroll
  for (int t = 0; t < 8; ++t)
#pragma unroll
    for (int r = 0; r < 8; ++r)
      P[(m0 + g * 8 + r) * WN + t * 16 + c] = f2bf(s[t][r]);
  __syncthreads();

  // O = P @ V   (K dim = 128 tokens, N dim = 32 padded head dim)
  v8f o0 = zero, o1 = zero;
  const bf16_t* prow = P + (size_t)(m0 + c) * WN;
#pragma unroll
  for (int kt = 0; kt < 4; ++kt) {
    int k0 = kt * 32;
    v16bf af = mkfrag(*(const v8bf*)(prow + k0 + 16 * g), *(const v8bf*)(prow + k0 + 16 * g + 8));
    v16bf b0 = *(const v16bf*)(Vs + (size_t)(c) * WN + k0 + 16 * g);
    v16bf b1 = *(const v16bf*)(Vs + (size_t)(16 + c) * WN + k0 + 16 * g);
    o0 = __builtin_amdgcn_wmma_f32_16x16x32_bf16(false, af, false, b0, (short)0, o0, false, false);
    o1 = __builtin_amdgcn_wmma_f32_16x16x32_bf16(false, af, false, b1, (short)0, o1, false, false);
  }

  // store (win, token, head*24 + d) into padded bf16 activation rows
#pragma unroll
  for (int r = 0; r < 8; ++r) {
    int m = m0 + g * 8 + r;
    ao[((size_t)(win * WN + m)) * KP + head * HD + c] = f2bf(o0[r]);
    int d1 = 16 + c;
    if (d1 < HD)
      ao[((size_t)(win * WN + m)) * KP + head * HD + d1] = f2bf(o1[r]);
  }
}

// ---------- proj epilogue: window-reverse (+unshift) + bias + residual add ----------
__global__ void proj_add_k(const float* __restrict__ po, const float* __restrict__ pb,
                           float* __restrict__ f, int shifted)
{
  int i = blockIdx.x * 256 + threadIdx.x;
  if (i >= TOK * CH) return;
  int tok = i / CH, ch = i % CH;
  int b = tok / (D_SZ * H_SZ * W_SZ);
  int r = tok % (D_SZ * H_SZ * W_SZ);
  int d = r / (H_SZ * W_SZ);
  int h = (r / W_SZ) % H_SZ;
  int w = r % W_SZ;
  int sd = d, sh = h, sw = w;
  if (shifted) { sd = (d + D_SZ - 1) % D_SZ; sh = (h + H_SZ - 4) % H_SZ; sw = (w + W_SZ - 4) % W_SZ; }
  int wdb = sd >> 1, zd = sd & 1, whb = sh >> 3, zy = sh & 7, wwb = sw >> 3, zx = sw & 7;
  int win = ((b * 3 + wdb) * 12 + whb) * 12 + wwb;
  int t = (zd * 8 + zy) * 8 + zx;
  f[i] += po[(size_t)(win * WN + t) * CH + ch] + pb[ch];
}

// ---------- f32 -> padded bf16 rows (no LN) ----------
__global__ void cvt_act_k(const float* __restrict__ f, bf16_t* __restrict__ out)
{
  int i = blockIdx.x * 256 + threadIdx.x;
  if (i >= TOK * KP) return;
  int tok = i / KP, ch = i % KP;
  out[i] = (ch < CH) ? f2bf(f[(size_t)tok * CH + ch]) : f2bf(0.f);
}

// ---------- weight f32 -> bf16 with K padding ----------
__global__ void cvt_w_k(const float* __restrict__ src, bf16_t* __restrict__ dst,
                        int rows, int K, int Kp)
{
  int i = blockIdx.x * 256 + threadIdx.x;
  if (i >= rows * Kp) return;
  int rr = i / Kp, cc = i % Kp;
  dst[i] = (cc < K) ? f2bf(src[(size_t)rr * K + cc]) : f2bf(0.f);
}

__global__ void zero_k(unsigned int* __restrict__ p, int n)
{
  int i = blockIdx.x * 256 + threadIdx.x;
  if (i < n) p[i] = 0u;
}

// ---------- trailing LN + transpose to (B, D, C, H, W) ----------
__global__ __launch_bounds__(256) void final_k(
    const float* __restrict__ y0, const float* __restrict__ g, const float* __restrict__ bb,
    float* __restrict__ out)
{
  int wave = threadIdx.x >> 5, lane = threadIdx.x & 31;
  int tok = blockIdx.x * 8 + wave;
  int b = tok / (D_SZ * H_SZ * W_SZ);
  int r = tok % (D_SZ * H_SZ * W_SZ);
  int d = r / (H_SZ * W_SZ);
  int h = (r / W_SZ) % H_SZ;
  int w = r % W_SZ;
  const float* row = y0 + (size_t)tok * CH;
  float vals[5]; float s = 0.f, ss = 0.f;
#pragma unroll
  for (int it = 0; it < 5; ++it) {
    int ch = lane + it * 32; float v = 0.f;
    if (ch < CH) { v = row[ch]; s += v; ss += v * v; }
    vals[it] = v;
  }
  for (int m = 16; m >= 1; m >>= 1) { s += __shfl_xor(s, m, 32); ss += __shfl_xor(ss, m, 32); }
  float mu = s / CH;
  float rs = rsqrtf(ss / CH - mu * mu + 1e-5f);
#pragma unroll
  for (int it = 0; it < 5; ++it) {
    int ch = lane + it * 32;
    if (ch < CH) {
      float o = (vals[it] - mu) * rs * g[ch] + bb[ch];
      out[((((size_t)b * D_SZ + d) * CH + ch) * H_SZ + h) * W_SZ + w] = o;
    }
  }
}

// ================= launcher =================
extern "C" void kernel_launch(void* const* d_in, const int* in_sizes, int n_in,
                              void* d_out, int out_size, void* d_ws, size_t ws_size,
                              hipStream_t stream)
{
  (void)in_sizes; (void)n_in; (void)out_size; (void)ws_size;
  const float* X      = (const float*)d_in[0];
  const float* conv_w = (const float*)d_in[1];
  const float* conv_b = (const float*)d_in[2];
  const float* ln0_g  = (const float*)d_in[3];
  const float* ln0_b  = (const float*)d_in[4];
  const float* n1g    = (const float*)d_in[5];
  const float* n1b    = (const float*)d_in[6];
  const float* qkvw   = (const float*)d_in[7];
  const float* qkvb   = (const float*)d_in[8];
  const float* rpb    = (const float*)d_in[9];
  const float* projw  = (const float*)d_in[10];
  const float* projb  = (const float*)d_in[11];
  const float* n2g    = (const float*)d_in[12];
  const float* n2b    = (const float*)d_in[13];
  const float* fc1w   = (const float*)d_in[14];
  const float* fc1b   = (const float*)d_in[15];
  const float* fc2w   = (const float*)d_in[16];
  const float* fc2b   = (const float*)d_in[17];
  const float* lin_w  = (const float*)d_in[18];
  const float* lin_b  = (const float*)d_in[19];
  const float* ln1_g  = (const float*)d_in[20];
  const float* ln1_b  = (const float*)d_in[21];

  // ---- carve workspace ----
  size_t off = 0;
  char* base = (char*)d_ws;
  auto carve = [&](size_t bytes) {
    void* p = base + off;
    off += (bytes + 255) & ~(size_t)255;
    return p;
  };
  float*  f    = (float*)carve((size_t)TOK * CH * 4);
  float*  y0   = (float*)carve((size_t)TOK * CH * 4);
  bf16_t* act  = (bf16_t*)carve((size_t)TOK * KP * 2);        // xw / ao / z / fb staging
  float*  go   = (float*)carve((size_t)TOK * CH * 4);         // proj f32 output
  bf16_t* qb   = (bf16_t*)carve((size_t)NWIN * HEADS * WN * HDP * 2);
  bf16_t* kb   = (bf16_t*)carve((size_t)NWIN * HEADS * WN * HDP * 2);
  bf16_t* vb   = (bf16_t*)carve((size_t)NWIN * HEADS * WN * HDP * 2);
  bf16_t* hid  = qb;                                          // alias: q/k dead during MLP
  bf16_t* wqkv = (bf16_t*)carve((size_t)2 * 432 * KP * 2);
  bf16_t* wprj = (bf16_t*)carve((size_t)2 * CH * KP * 2);
  bf16_t* wfc1 = (bf16_t*)carve((size_t)2 * HID * KP * 2);
  bf16_t* wfc2 = (bf16_t*)carve((size_t)2 * CH * HID * 2);
  bf16_t* wlin = (bf16_t*)carve((size_t)CH * KP * 2);

  // ---- weights -> bf16 (K-padded) ----
  auto cvtw = [&](const float* src, bf16_t* dst, int rows, int K, int Kp) {
    int n = rows * Kp;
    cvt_w_k<<<(n + 255) / 256, 256, 0, stream>>>(src, dst, rows, K, Kp);
  };
  cvtw(qkvw, wqkv, 2 * 432, CH, KP);
  cvtw(projw, wprj, 2 * CH, CH, KP);
  cvtw(fc1w, wfc1, 2 * HID, CH, KP);
  cvtw(fc2w, wfc2, 2 * CH, HID, HID);
  cvtw(lin_w, wlin, CH, CH, KP);

  // ---- conv + LN0 ----
  conv_ln0_k<<<TOK / 8, 256, 0, stream>>>(X, conv_w, conv_b, ln0_g, ln0_b, y0, f);

  const float scale = 0.20412414523193154f;  // 24^-0.5

  for (int i = 0; i < 2; ++i) {
    int shifted = i & 1;
    // LN1 + (shift) + window partition -> bf16
    ln_part_k<<<TOK / 8, 256, 0, stream>>>(f, n1g + i * CH, n1b + i * CH, act, 1, shifted);
    // zero q/k/vT (head-dim padding must be zero)
    int zn = NWIN * HEADS * WN * HDP * 3 / 2;  // as uint32
    zero_k<<<(zn + 255) / 256, 256, 0, stream>>>((unsigned int*)qb, zn);
    // qkv GEMM, fused split/scale/transpose
    gemm_wmma_k<<<dim3(TOK / 128, 432 / 48), 256, 0, stream>>>(
        act, KP, wqkv + (size_t)i * 432 * KP, KP, 432, qkvb + i * 432,
        1, scale, nullptr, 0, nullptr, 0, qb, kb, vb);
    // attention
    attn_k<<<NWIN * HEADS, 256, 0, stream>>>(qb, kb, vb, rpb + (size_t)i * TBL * HEADS, act, shifted);
    // proj GEMM -> f32
    gemm_wmma_k<<<dim3(TOK / 128, CH / 48), 256, 0, stream>>>(
        act, KP, wprj + (size_t)i * CH * KP, KP, CH, nullptr,
        0, 0.f, go, CH, nullptr, 0, nullptr, nullptr, nullptr);
    // window-reverse + unshift + bias + residual
    proj_add_k<<<(TOK * CH + 255) / 256, 256, 0, stream>>>(go, projb + i * CH, f, shifted);
    // LN2 -> bf16 (token-major)
    ln_part_k<<<TOK / 8, 256, 0, stream>>>(f, n2g + i * CH, n2b + i * CH, act, 0, 0);
    // fc1 GEMM + GeLU -> bf16 hidden
    gemm_wmma_k<<<dim3(TOK / 128, HID / 48), 256, 0, stream>>>(
        act, KP, wfc1 + (size_t)i * HID * KP, KP, HID, fc1b + i * HID,
        2, 0.f, nullptr, 0, hid, HID, nullptr, nullptr, nullptr);
    // fc2 GEMM + bias + residual add into f
    gemm_wmma_k<<<dim3(TOK / 128, CH / 48), 256, 0, stream>>>(
        hid, HID, wfc2 + (size_t)i * CH * HID, HID, CH, fc2b + i * CH,
        3, 0.f, f, CH, nullptr, 0, nullptr, nullptr, nullptr);
  }

  // final linear + skip into y0
  cvt_act_k<<<(TOK * KP + 255) / 256, 256, 0, stream>>>(f, act);
  gemm_wmma_k<<<dim3(TOK / 128, CH / 48), 256, 0, stream>>>(
      act, KP, wlin, KP, CH, lin_b,
      3, 0.f, y0, CH, nullptr, 0, nullptr, nullptr, nullptr);

  // trailing LN + layout transpose
  final_k<<<TOK / 8, 256, 0, stream>>>(y0, ln1_g, ln1_b, (float*)d_out);
}
